// Model_44753559225046
// MI455X (gfx1250) — matrix operand order repaired and checked by reading, not compiled, not run on hardware
//
#include <hip/hip_runtime.h>
#include <math.h>

// ---------------- model dims ----------------
#define NB   64      // batch
#define NS   512     // input seq len
#define ND   128     // d_model
#define NCIN 7       // input channels
#define NL   1024    // S + P (pred length)
#define NF   512     // DFT bins computed (true freq = bin index + 1)
#define TPAD 2048    // worst-case padded time for period reshape temps
#define GM   32      // M rows per GEMM block (2 WMMA tiles per wave)

typedef _Float16 h8   __attribute__((ext_vector_type(8)));
typedef _Float16 v16h __attribute__((ext_vector_type(16)));
typedef float    v8f  __attribute__((ext_vector_type(8)));

// ---------------- WMMA 16x16x32 f16 fragment loaders (wave32) ----------------
// A (16M x 32K): lane l<16 holds row l with K {0..7 | 16..23}, lanes 16..31 the
// +8 K halves -> two contiguous 8-half (16B) runs per lane.
__device__ __forceinline__ v16h frag_a(const _Float16* __restrict__ A, int K,
                                       int m0, int k0, int lane) {
  int lr = lane & 15, hf = lane >> 4;
  const _Float16* p = A + (size_t)(m0 + lr) * K + k0 + hf * 8;
  h8 lo = *(const h8*)p;
  h8 hi = *(const h8*)(p + 16);
  return __builtin_shufflevector(lo, hi, 0,1,2,3,4,5,6,7,8,9,10,11,12,13,14,15);
}
// B (32K x 16N) sourced from an N-major (N x K) buffer: lane l<16 holds column
// l with K {0..7 | 16..23}, lanes 16..31 the +8 K halves (matches frag_a).
__device__ __forceinline__ v16h frag_b(const _Float16* __restrict__ Bt, int K,
                                       int n0, int k0, int lane) {
  int lr = lane & 15, hf = lane >> 4;
  const _Float16* p = Bt + (size_t)(n0 + lr) * K + k0 + hf * 8;
  h8 lo = *(const h8*)p;
  h8 hi = *(const h8*)(p + 16);
  return __builtin_shufflevector(lo, hi, 0,1,2,3,4,5,6,7,8,9,10,11,12,13,14,15);
}

// ---------------- utility kernels ----------------
__global__ void k_zero(float* __restrict__ p, int n) {
  int i = blockIdx.x * 256 + threadIdx.x;
  if (i < n) p[i] = 0.f;
}

__global__ void k_cvt_h(const float* __restrict__ s, _Float16* __restrict__ d, int n) {
  int i = blockIdx.x * 256 + threadIdx.x;
  if (i < n) d[i] = (_Float16)s[i];
}

// per-(b,c) mean / stdev over time (RevIN-style normalization stats)
__global__ void k_stats(const float* __restrict__ x, float* __restrict__ mean,
                        float* __restrict__ stdev) {
  int b = blockIdx.x / NCIN, c = blockIdx.x % NCIN;
  float s = 0.f, s2 = 0.f;
  for (int t = threadIdx.x; t < NS; t += 128) {
    float v = x[((size_t)b * NS + t) * NCIN + c];
    s += v; s2 += v * v;
  }
  __shared__ float sh[2][4];
  for (int off = 16; off; off >>= 1) { s += __shfl_down(s, off); s2 += __shfl_down(s2, off); }
  int lane = threadIdx.x & 31, w = threadIdx.x >> 5;
  if (!lane) { sh[0][w] = s; sh[1][w] = s2; }
  __syncthreads();
  if (threadIdx.x == 0) {
    float S1 = sh[0][0] + sh[0][1] + sh[0][2] + sh[0][3];
    float S2 = sh[1][0] + sh[1][1] + sh[1][2] + sh[1][3];
    float m = S1 / (float)NS;
    float var = S2 / (float)NS - m * m;
    mean[blockIdx.x]  = m;
    stdev[blockIdx.x] = sqrtf(var + 1e-5f);
  }
}

// ---------------- embedding (circular conv1d + pos-embed + time-mark) ----------------
__device__ __forceinline__ float pos_embed(int s, int d) {
  float div = __expf(-(float)(d & ~1) * (9.210340371976184f / 128.f)); // ln(1e4)/D
  float ang = (float)s * div;
  return (d & 1) ? __cosf(ang) : __sinf(ang);
}

// single (last) channel variant -> writes f16 N-major (b, d, s) for WMMA B
__global__ void k_embed1(const float* __restrict__ x, const float* __restrict__ mark,
                         const float* __restrict__ wt, const float* __restrict__ wtime,
                         const float* __restrict__ mean, const float* __restrict__ stdev,
                         _Float16* __restrict__ eT) {
  int idx = blockIdx.x * 256 + threadIdx.x;      // b * NS * ND
  int d = idx & 127, s = (idx >> 7) & 511, b = idx >> 16;
  int sm = (s + 511) & 511, sp = (s + 1) & 511;  // circular
  float mu = mean[b * NCIN + 6], inv = 1.f / stdev[b * NCIN + 6];
  float xm = (x[((size_t)b * NS + sm) * NCIN + 6] - mu) * inv;
  float x0 = (x[((size_t)b * NS + s ) * NCIN + 6] - mu) * inv;
  float xp = (x[((size_t)b * NS + sp) * NCIN + 6] - mu) * inv;
  const float* wr = wt + d * 3;
  float acc = wr[0] * xm + wr[1] * x0 + wr[2] * xp;
  acc += pos_embed(s, d);
  const float* mk = mark + ((size_t)b * NS + s) * 4;
  acc += mk[0] * wtime[0 * ND + d] + mk[1] * wtime[1 * ND + d] +
         mk[2] * wtime[2 * ND + d] + mk[3] * wtime[3 * ND + d];
  eT[((size_t)b * ND + d) * NS + s] = (_Float16)acc;
}

// full CIN-channel variant
__global__ void k_embedF(const float* __restrict__ x, const float* __restrict__ mark,
                         const float* __restrict__ wt, const float* __restrict__ wtime,
                         const float* __restrict__ mean, const float* __restrict__ stdev,
                         _Float16* __restrict__ eT) {
  int idx = blockIdx.x * 256 + threadIdx.x;
  int d = idx & 127, s = (idx >> 7) & 511, b = idx >> 16;
  int sm = (s + 511) & 511, sp = (s + 1) & 511;
  float acc = 0.f;
  for (int c = 0; c < NCIN; ++c) {
    float mu = mean[b * NCIN + c], inv = 1.f / stdev[b * NCIN + c];
    float xm = (x[((size_t)b * NS + sm) * NCIN + c] - mu) * inv;
    float x0 = (x[((size_t)b * NS + s ) * NCIN + c] - mu) * inv;
    float xp = (x[((size_t)b * NS + sp) * NCIN + c] - mu) * inv;
    const float* wr = wt + (d * NCIN + c) * 3;
    acc += wr[0] * xm + wr[1] * x0 + wr[2] * xp;
  }
  acc += pos_embed(s, d);
  const float* mk = mark + ((size_t)b * NS + s) * 4;
  acc += mk[0] * wtime[0 * ND + d] + mk[1] * wtime[1 * ND + d] +
         mk[2] * wtime[2 * ND + d] + mk[3] * wtime[3 * ND + d];
  eT[((size_t)b * ND + d) * NS + s] = (_Float16)acc;
}

// ---------------- WMMA GEMM: C[b,m,n] = sum_k A[m,k] * Bt[b,n,k] + bias[m] ----------------
// block = 256 threads = 8 waves, grid = (M/32, batch).
// The 32 x K A-tile (rows m0..m0+31 of row-major A, i.e. one contiguous 32KB
// slab) is DMA'd into LDS with GLOBAL_LOAD_ASYNC_TO_LDS_B128 (ASYNCcnt), then
// all 8 waves read A fragments from LDS (ds_load_b128) while streaming their
// private 16-wide B panel from global. 2 WMMA tiles per wave reuse each B frag.
__global__ void k_gemm_f16(const _Float16* __restrict__ A, const _Float16* __restrict__ Bt,
                           const float* __restrict__ bias, float* __restrict__ C,
                           int M, int N, int K) {
  __shared__ __align__(16) _Float16 shA[GM * NS];   // K <= 512 -> 32KB
  int t = threadIdx.x;
  int lane = t & 31, wave = t >> 5;
  int m0 = blockIdx.x * GM, n0 = wave * 16, b = blockIdx.y;

  // ---- async DMA of the contiguous A slab into LDS ----
  {
    const char* g = (const char*)(A + (size_t)m0 * K) + t * 16;
    // low 32 bits of the generic shared-memory address are the wave-relative
    // LDS byte offset
    unsigned l = (unsigned)(size_t)shA + (unsigned)(t * 16);
    int iters = (GM * K * 2) / (256 * 16);          // 8 for K=512
    for (int i = 0; i < iters; ++i) {
      asm volatile("global_load_async_to_lds_b128 %0, %1, off"
                   :: "v"(l), "v"(g) : "memory");
      l += 4096;
      g += 4096;
    }
    asm volatile("s_wait_asynccnt 0" ::: "memory");
  }
  __syncthreads();

  const _Float16* Bb = Bt + (size_t)b * N * K;
  const _Float16* pB = Bb + (size_t)(n0 + (lane & 15)) * K;
  v8f acc0 = {}, acc1 = {};
  for (int k0 = 0; k0 < K; k0 += 32) {
    __builtin_prefetch(pB + k0 + 512, 0, 1);        // stream-ahead into GL2
    v16h bf = frag_b(Bb, K, n0, k0, lane);
    v16h a0 = frag_a(shA, K, 0,  k0, lane);         // ds_load_b128 x2
    v16h a1 = frag_a(shA, K, 16, k0, lane);
    acc0 = __builtin_amdgcn_wmma_f32_16x16x32_f16(false, a0, false, bf,
                                                  (short)0, acc0, false, false);
    acc1 = __builtin_amdgcn_wmma_f32_16x16x32_f16(false, a1, false, bf,
                                                  (short)0, acc1, false, false);
  }
  int lr = lane & 15, hf = lane >> 4;
  float* Cb = C + (size_t)b * M * N;
#pragma unroll
  for (int r = 0; r < 8; ++r) {
    int m = m0 + r + hf * 8;
    Cb[(size_t)m * N + n0 + lr]        = acc0[r] + bias[m];
    Cb[(size_t)(m + 16) * N + n0 + lr] = acc1[r] + bias[m + 16];
  }
}

// ---------------- DFT amplitude via WMMA (rfft bins 1..512) ----------------
__global__ void k_dft_basis(_Float16* __restrict__ Cb, _Float16* __restrict__ Sb) {
  int idx = blockIdx.x * 256 + threadIdx.x;   // NF * NL
  int t = idx & 1023, f = idx >> 10;
  int ph = ((f + 1) * t) & 1023;              // mod NL range reduction
  float ang = 6.283185307179586f * (float)ph / 1024.f;
  Cb[idx] = (_Float16)__cosf(ang);
  Sb[idx] = (_Float16)__sinf(ang);
}

// transpose-convert enc (b,t,n) f32 -> (b,n,t) f16 for the DFT B operand
__global__ void k_cvt_encT(const float* __restrict__ e, _Float16* __restrict__ eT) {
  int idx = blockIdx.x * 256 + threadIdx.x;   // b * NL * ND
  int n = idx & 127, t = (idx >> 7) & 1023, b = idx >> 17;
  eT[((size_t)b * ND + n) * NL + t] = (_Float16)e[idx];
}

// Re/Im = basis GEMMs; |X| reduced over channels into amp[b, f] (sum over 128 n)
__global__ void k_dft_amp(const _Float16* __restrict__ Cb, const _Float16* __restrict__ Sb,
                          const _Float16* __restrict__ Xt, float* __restrict__ amp) {
  int lane = threadIdx.x & 31, wave = threadIdx.x >> 5;
  int f0 = blockIdx.x * 16, n0 = wave * 16, b = blockIdx.y;
  const _Float16* Bb = Xt + (size_t)b * ND * NL;
  v8f cre = {}, cim = {};
  for (int k0 = 0; k0 < NL; k0 += 32) {
    v16h bf = frag_b(Bb, NL, n0, k0, lane);
    v16h ac = frag_a(Cb, NL, f0, k0, lane);
    cre = __builtin_amdgcn_wmma_f32_16x16x32_f16(false, ac, false, bf,
                                                 (short)0, cre, false, false);
    v16h as = frag_a(Sb, NL, f0, k0, lane);
    cim = __builtin_amdgcn_wmma_f32_16x16x32_f16(false, as, false, bf,
                                                 (short)0, cim, false, false);
  }
  int lr = lane & 15, hf = lane >> 4;
#pragma unroll
  for (int r = 0; r < 8; ++r) {
    float a = sqrtf(cre[r] * cre[r] + cim[r] * cim[r]);
    a += __shfl_xor(a, 1); a += __shfl_xor(a, 2);
    a += __shfl_xor(a, 4); a += __shfl_xor(a, 8);   // sum 16 n-lanes
    if (lr == 0) atomicAdd(&amp[(size_t)b * NF + f0 + r + hf * 8], a);
  }
}

// top-2 frequencies (mean amp over batch+channels) + per-batch softmax weights
__global__ void k_topk(const float* __restrict__ amp, int* __restrict__ meta,
                       float* __restrict__ wsoft) {
  __shared__ float freq[NF];
  __shared__ int si0, si1;
  int f = threadIdx.x;                 // 512 threads
  float s = 0.f;
  for (int b = 0; b < NB; ++b) s += amp[b * NF + f];
  freq[f] = s;
  __syncthreads();
  if (f == 0) {
    int i0 = 0; float v0 = -1.f;
    for (int i = 0; i < NF; ++i) if (freq[i] > v0) { v0 = freq[i]; i0 = i; }
    int i1 = (i0 == 0) ? 1 : 0; float v1 = freq[i1];
    for (int i = 0; i < NF; ++i) if (i != i0 && freq[i] > v1) { v1 = freq[i]; i1 = i; }
    si0 = i0; si1 = i1;
    int fr0 = i0 + 1, fr1 = i1 + 1;
    int p0 = NL / fr0, p1 = NL / fr1;
    meta[0] = fr0; meta[1] = fr1;
    meta[2] = p0;  meta[3] = p1;
    meta[4] = ((NL + p0 - 1) / p0) * p0;
    meta[5] = ((NL + p1 - 1) / p1) * p1;
  }
  __syncthreads();
  if (f < NB) {
    float a0 = amp[f * NF + si0] * (1.f / (float)ND);
    float a1 = amp[f * NF + si1] * (1.f / (float)ND);
    float mx = fmaxf(a0, a1);
    float e0 = __expf(a0 - mx), e1 = __expf(a1 - mx);
    wsoft[f * 2 + 0] = e0 / (e0 + e1);
    wsoft[f * 2 + 1] = e1 / (e0 + e1);
  }
}

// depthwise 3x3 SAME conv + relu on the period-reshaped (H=len/p, W=p) images,
// one image per (b, n). Geometry is data-dependent -> read from meta, static grid.
__global__ void k_dwconv_times(const float* __restrict__ encSrc, const float* __restrict__ tmpSrc,
                               float* __restrict__ dst, const int* __restrict__ meta,
                               int slot, int fromEnc,
                               const float* __restrict__ w9, const float* __restrict__ bptr) {
  int idx = blockIdx.x * 256 + threadIdx.x;  // b * TPAD * ND
  int n = idx & 127, t = (idx >> 7) & 2047, b = idx >> 18;
  int p = meta[2 + slot], len = meta[4 + slot];
  if (t >= len) return;
  int H = len / p;
  int h = t / p, w = t - h * p;
  float acc = 0.f;
#pragma unroll
  for (int dh = -1; dh <= 1; ++dh) {
    int hh = h + dh;
    if ((unsigned)hh >= (unsigned)H) continue;
#pragma unroll
    for (int dw = -1; dw <= 1; ++dw) {
      int ww = w + dw;
      if ((unsigned)ww >= (unsigned)p) continue;
      int tt = hh * p + ww;
      float v;
      if (fromEnc) v = (tt < NL) ? encSrc[((size_t)b * NL + tt) * ND + n] : 0.f;
      else         v = tmpSrc[((size_t)b * TPAD + tt) * ND + n];
      acc += w9[(dh + 1) * 3 + (dw + 1)] * v;
    }
  }
  dst[((size_t)b * TPAD + t) * ND + n] = fmaxf(acc + bptr[0], 0.f);
}

// encOut += softmax_weight[b, slot] * conv_result (truncated back to NL)
__global__ void k_times_accum(const float* __restrict__ tmp, const float* __restrict__ wsoft,
                              float* __restrict__ out, int slot) {
  int idx = blockIdx.x * 256 + threadIdx.x;  // b * NL * ND
  int n = idx & 127, t = (idx >> 7) & 1023, b = idx >> 17;
  out[idx] += wsoft[b * 2 + slot] * tmp[((size_t)b * TPAD + t) * ND + n];
}

// depthwise 3x3 SAME conv + relu on the (NL x ND) single-channel images (enc1 path)
__global__ void k_dwconv_img(const float* __restrict__ src, float* __restrict__ dst,
                             const float* __restrict__ w9, const float* __restrict__ bptr) {
  int idx = blockIdx.x * 256 + threadIdx.x;  // b * NL * ND
  int w = idx & 127, h = (idx >> 7) & 1023, b = idx >> 17;
  float acc = 0.f;
#pragma unroll
  for (int dh = -1; dh <= 1; ++dh) {
    int hh = h + dh;
    if ((unsigned)hh >= (unsigned)NL) continue;
#pragma unroll
    for (int dw = -1; dw <= 1; ++dw) {
      int ww = w + dw;
      if ((unsigned)ww >= (unsigned)ND) continue;
      acc += w9[(dh + 1) * 3 + (dw + 1)] * src[((size_t)b * NL + hh) * ND + ww];
    }
  }
  dst[idx] = fmaxf(acc + bptr[0], 0.f);
}

// dec[b,l] = enc[b,l,:] . w_proj + b_proj
__global__ void k_proj(const float* __restrict__ src, const float* __restrict__ wp,
                       const float* __restrict__ bp, float* __restrict__ dst) {
  int idx = blockIdx.x * 256 + threadIdx.x;  // b * NL
  const float* r = src + (size_t)idx * ND;
  float acc = 0.f;
#pragma unroll 8
  for (int d = 0; d < ND; ++d) acc += r[d] * wp[d];
  dst[idx] = acc + bp[0];
}

// CBAM channel attention: ca = sigmoid(mlp(mean_L) + mlp(max_L)), mlp: 2->1->2
__global__ void k_cbam_stats(const float* __restrict__ dec, const float* __restrict__ dec1,
                             const float* __restrict__ w1, const float* __restrict__ w2,
                             float* __restrict__ ca) {
  int b = blockIdx.x;
  float s0 = 0.f, s1 = 0.f, m0 = -1e30f, m1 = -1e30f;
  for (int l = threadIdx.x; l < NL; l += 256) {
    float v0 = dec[b * NL + l], v1 = dec1[b * NL + l];
    s0 += v0; s1 += v1; m0 = fmaxf(m0, v0); m1 = fmaxf(m1, v1);
  }
  __shared__ float sh[4][8];
  for (int off = 16; off; off >>= 1) {
    s0 += __shfl_down(s0, off); s1 += __shfl_down(s1, off);
    m0 = fmaxf(m0, __shfl_down(m0, off)); m1 = fmaxf(m1, __shfl_down(m1, off));
  }
  int lane = threadIdx.x & 31, w = threadIdx.x >> 5;
  if (!lane) { sh[0][w] = s0; sh[1][w] = s1; sh[2][w] = m0; sh[3][w] = m1; }
  __syncthreads();
  if (threadIdx.x == 0) {
    s0 = s1 = 0.f; m0 = m1 = -1e30f;
    for (int i = 0; i < 8; ++i) {
      s0 += sh[0][i]; s1 += sh[1][i];
      m0 = fmaxf(m0, sh[2][i]); m1 = fmaxf(m1, sh[3][i]);
    }
    float a0 = s0 / (float)NL, a1 = s1 / (float)NL;
    float mavg = fmaxf(a0 * w1[0] + a1 * w1[1], 0.f);
    float mmax = fmaxf(m0 * w1[0] + m1 * w1[1], 0.f);
    float c0 = (mavg + mmax) * w2[0];
    float c1 = (mavg + mmax) * w2[1];
    ca[b * 2 + 0] = 1.f / (1.f + __expf(-c0));
    ca[b * 2 + 1] = 1.f / (1.f + __expf(-c1));
  }
}

// spatial-attention inputs: sbuf[b,0,l]=max_c(out_c), sbuf[b,1,l]=mean_c(out_c)
__global__ void k_cbam_s(const float* __restrict__ dec, const float* __restrict__ dec1,
                         const float* __restrict__ ca, float* __restrict__ sbuf) {
  int idx = blockIdx.x * 256 + threadIdx.x;  // b * NL
  int l = idx & 1023, b = idx >> 10;
  float o0 = dec[idx] * ca[b * 2], o1 = dec1[idx] * ca[b * 2 + 1];
  sbuf[(size_t)b * 2 * NL + l]      = fmaxf(o0, o1);
  sbuf[(size_t)b * 2 * NL + NL + l] = 0.5f * (o0 + o1);
}

// big (1024,1) SAME spatial conv -> sa, residual, 2->1 head mlp, denormalize
__global__ void k_final(const float* __restrict__ dec, const float* __restrict__ dec1,
                        const float* __restrict__ ca, const float* __restrict__ sbuf,
                        const float* __restrict__ wsp, const float* __restrict__ bsp,
                        const float* __restrict__ wmlp, const float* __restrict__ bmlp,
                        const float* __restrict__ mean, const float* __restrict__ stdev,
                        float* __restrict__ out) {
  int idx = blockIdx.x * 256 + threadIdx.x;  // b * NL
  int l = idx & 1023, b = idx >> 10;
  const float* s0 = sbuf + (size_t)b * 2 * NL;
  const float* s1 = s0 + NL;
  float acc = bsp[0];
  for (int k = 0; k < NL; ++k) {           // SAME pad: lo=511, hi=512
    int pos = l + k - 511;
    if ((unsigned)pos < (unsigned)NL)
      acc += s0[pos] * wsp[k] + s1[pos] * wsp[NL + k];
  }
  float sa = 1.f / (1.f + __expf(-acc));
  float x0 = dec[idx], x1 = dec1[idx];
  float o0 = x0 * ca[b * 2], o1 = x1 * ca[b * 2 + 1];
  float d0 = o0 * sa + x0, d1 = o1 * sa + x1;
  float y = d0 * wmlp[0] + d1 * wmlp[1] + bmlp[0];
  out[idx] = y * stdev[b * NCIN + 6] + mean[b * NCIN + 6];
}

// ---------------- host orchestration ----------------
extern "C" void kernel_launch(void* const* d_in, const int* in_sizes, int n_in,
                              void* d_out, int out_size, void* d_ws, size_t ws_size,
                              hipStream_t stream) {
  (void)in_sizes; (void)n_in; (void)out_size; (void)ws_size;
  const float* x      = (const float*)d_in[0];
  const float* mark   = (const float*)d_in[1];
  const float* wtok1  = (const float*)d_in[2];
  const float* wtok   = (const float*)d_in[3];
  const float* wtime1 = (const float*)d_in[4];
  const float* wtime  = (const float*)d_in[5];
  const float* wpred  = (const float*)d_in[6];
  const float* bpred  = (const float*)d_in[7];
  const float* tbw    = (const float*)d_in[8];
  const float* tbb    = (const float*)d_in[9];
  const float* convw  = (const float*)d_in[10];
  const float* convb  = (const float*)d_in[11];
  const float* wproj  = (const float*)d_in[12];
  const float* bproj  = (const float*)d_in[13];
  const float* wca1   = (const float*)d_in[14];
  const float* wca2   = (const float*)d_in[15];
  const float* wsp    = (const float*)d_in[16];
  const float* bsp    = (const float*)d_in[17];
  const float* wmlp   = (const float*)d_in[18];
  const float* bmlp   = (const float*)d_in[19];
  float* out = (float*)d_out;

  char* ws = (char*)d_ws;
  size_t off = 0;
  auto take = [&](size_t bytes) -> char* {
    size_t a = (off + 255) & ~(size_t)255;
    off = a + bytes;
    return ws + a;
  };
  float*    mean   = (float*)take((size_t)NB * NCIN * 4);
  float*    stdev  = (float*)take((size_t)NB * NCIN * 4);
  _Float16* wpredH = (_Float16*)take((size_t)NL * NS * 2);
  _Float16* eT     = (_Float16*)take((size_t)NB * ND * NS * 2);
  float*    encA   = (float*)take((size_t)NB * NL * ND * 4);
  float*    encB   = (float*)take((size_t)NB * NL * ND * 4);
  _Float16* encT   = (_Float16*)take((size_t)NB * ND * NL * 2);
  _Float16* dftC   = (_Float16*)take((size_t)NF * NL * 2);
  _Float16* dftS   = (_Float16*)take((size_t)NF * NL * 2);
  float*    ampbf  = (float*)take((size_t)NB * NF * 4);
  int*      meta   = (int*)take(256);
  float*    wsoft  = (float*)take((size_t)NB * 2 * 4);
  float*    tmpA   = (float*)take((size_t)NB * TPAD * ND * 4);
  float*    tmpB   = (float*)take((size_t)NB * TPAD * ND * 4);
  float*    dec    = (float*)take((size_t)NB * NL * 4);
  float*    dec1   = (float*)take((size_t)NB * NL * 4);
  float*    ca     = (float*)take((size_t)NB * 2 * 4);
  float*    sbuf   = (float*)take((size_t)NB * 2 * NL * 4);

  const int g_emb  = (NB * NS * ND) / 256;   // 16384
  const int g_enc  = (NB * NL * ND) / 256;   // 32768
  const int g_pad  = (NB * TPAD * ND) / 256; // 65536
  const int g_bl   = (NB * NL) / 256;        // 256

  // --- preprocessing ---
  k_stats<<<NB * NCIN, 128, 0, stream>>>(x, mean, stdev);
  k_cvt_h<<<(NL * NS) / 256, 256, 0, stream>>>(wpred, wpredH, NL * NS);
  k_dft_basis<<<(NF * NL) / 256, 256, 0, stream>>>(dftC, dftS);

  // --- enc = pred_lin(embed(xe[:,:,-1:])) : WMMA GEMM, M=1024 K=512 N=128 ---
  k_embed1<<<g_emb, 256, 0, stream>>>(x, mark, wtok1, wtime1, mean, stdev, eT);
  k_gemm_f16<<<dim3(NL / GM, NB), 256, 0, stream>>>(wpredH, eT, bpred, encA, NL, ND, NS);

  // --- 2x TimesBlock (ping-pong encA <-> encB) ---
  float* cur = encA; float* nxt = encB;
  for (int li = 0; li < 2; ++li) {
    k_zero<<<(NB * NF + 255) / 256, 256, 0, stream>>>(ampbf, NB * NF);
    k_cvt_encT<<<g_enc, 256, 0, stream>>>(cur, encT);
    k_dft_amp<<<dim3(NF / 16, NB), 256, 0, stream>>>(dftC, dftS, encT, ampbf);
    k_topk<<<1, 512, 0, stream>>>(ampbf, meta, wsoft);
    k_zero<<<g_enc, 256, 0, stream>>>(nxt, NB * NL * ND);
    for (int slot = 0; slot < 2; ++slot) {
      k_dwconv_times<<<g_pad, 256, 0, stream>>>(cur, tmpB, tmpA, meta, slot, 1,
                                                tbw + (li * 3 + 0) * 9, tbb + li * 3 + 0);
      k_dwconv_times<<<g_pad, 256, 0, stream>>>(cur, tmpA, tmpB, meta, slot, 0,
                                                tbw + (li * 3 + 1) * 9, tbb + li * 3 + 1);
      k_dwconv_times<<<g_pad, 256, 0, stream>>>(cur, tmpB, tmpA, meta, slot, 0,
                                                tbw + (li * 3 + 2) * 9, tbb + li * 3 + 2);
      k_times_accum<<<g_enc, 256, 0, stream>>>(tmpA, wsoft, nxt, slot);
    }
    float* t = cur; cur = nxt; nxt = t;
  }
  // cur == encA (2 swaps): final encoder state
  k_proj<<<g_bl, 256, 0, stream>>>(cur, wproj, bproj, dec);

  // --- enc1 path: full-channel embed -> WMMA GEMM -> 3x depthwise img conv ---
  k_embedF<<<g_emb, 256, 0, stream>>>(x, mark, wtok, wtime, mean, stdev, eT);
  k_gemm_f16<<<dim3(NL / GM, NB), 256, 0, stream>>>(wpredH, eT, bpred, encB, NL, ND, NS);
  k_dwconv_img<<<g_enc, 256, 0, stream>>>(encB, tmpA, convw + 0, convb + 0);
  k_dwconv_img<<<g_enc, 256, 0, stream>>>(tmpA, tmpB, convw + 9, convb + 1);
  k_dwconv_img<<<g_enc, 256, 0, stream>>>(tmpB, tmpA, convw + 18, convb + 2);
  k_proj<<<g_bl, 256, 0, stream>>>(tmpA, wproj, bproj, dec1);

  // --- CBAM + head + denorm ---
  k_cbam_stats<<<NB, 256, 0, stream>>>(dec, dec1, wca1, wca2, ca);
  k_cbam_s<<<g_bl, 256, 0, stream>>>(dec, dec1, ca, sbuf);
  k_final<<<g_bl, 256, 0, stream>>>(dec, dec1, ca, sbuf, wsp, bsp, wmlp, bmlp,
                                    mean, stdev, out);
}
